// SwinV2Block_953482739890
// MI455X (gfx1250) — matrix operand order repaired and checked by reading, not compiled, not run on hardware
//
#include <hip/hip_runtime.h>
#include <cmath>

typedef __bf16 bf16_t;
typedef __attribute__((ext_vector_type(16))) __bf16 v16bf;
typedef __attribute__((ext_vector_type(8)))  __bf16 v8bf;
typedef __attribute__((ext_vector_type(8)))  float  v8f;

#define HEADS_ 3
#define D_     32
#define C_     96
#define NTOK   49
#define HW_    56
#define HID_   384

__device__ inline v8f zero8() {
  v8f z;
#pragma unroll
  for (int i = 0; i < 8; ++i) z[i] = 0.f;
  return z;
}

__device__ inline v8f wmma_bf16(v16bf a, v16bf b, v8f c) {
  return __builtin_amdgcn_wmma_f32_16x16x32_bf16(false, a, false, b, (short)0, c, false, false);
}

// A fragment (16x32 bf16, MxK): lane m = base+(lane&15); K = 16 contiguous at (lane>>4)*16
__device__ inline v16bf load_fragA(const bf16_t* base, int row, int stride, int kOff, int lane) {
  int r  = row + (lane & 15);
  int kb = (lane >> 4) << 4;
  return *(const v16bf*)(base + r * stride + kOff + kb);
}

// B fragment (32x16 bf16, KxN) read from W[n][k] row-major: lane n = base+(lane&15); K = 16 contiguous at (lane>>4)*16
__device__ inline v16bf load_fragB(const bf16_t* base, int nrow, int stride, int kOff, int lane) {
  int n  = nrow + (lane & 15);
  int ks = (lane >> 4) << 4;
  return *(const v16bf*)(base + n * stride + kOff + ks);
}

__device__ inline float gelu_exact(float x) {
  return 0.5f * x * (1.f + erff(x * 0.70710678118654752f));
}

// ---------------- weight f32 -> bf16 conversion ----------------
__global__ void convert_weights_kernel(const float* __restrict__ qkv_w,
                                       const float* __restrict__ proj_w,
                                       const float* __restrict__ fc1_w,
                                       const float* __restrict__ fc2_w,
                                       bf16_t* __restrict__ wq, bf16_t* __restrict__ wp,
                                       bf16_t* __restrict__ w1, bf16_t* __restrict__ w2) {
  int i = blockIdx.x * blockDim.x + threadIdx.x;
  int stride = gridDim.x * blockDim.x;
  for (int j = i; j < 288 * 96; j += stride) wq[j] = (bf16_t)qkv_w[j];
  for (int j = i; j < 96 * 96; j += stride)  wp[j] = (bf16_t)proj_w[j];
  for (int j = i; j < 384 * 96; j += stride) w1[j] = (bf16_t)fc1_w[j];
  for (int j = i; j < 96 * 384; j += stride) w2[j] = (bf16_t)fc2_w[j];
}

// ---------------- continuous position bias: 16*sigmoid(cpb_mlp(table))[rpi] ----------------
__device__ inline float cpb_coord(int a) {
  float t = (float)a * (8.0f / 6.0f);
  float l = log2f(fabsf(t) + 1.f) * (1.f / 3.f);
  return (t > 0.f) ? l : ((t < 0.f) ? -l : 0.f);
}

__global__ __launch_bounds__(256) void cpb_bias_kernel(const float* __restrict__ w1,
                                                       const float* __restrict__ b1,
                                                       const float* __restrict__ w2,
                                                       float* __restrict__ bias16) {
  __shared__ float tab[169 * 3];
  int tid = threadIdx.x;
  for (int e = tid; e < 169; e += 256) {
    int i = e / 13, j = e % 13;
    float tx = cpb_coord(i - 6), ty = cpb_coord(j - 6);
    float a0 = 0.f, a1 = 0.f, a2 = 0.f;
    for (int k = 0; k < 512; ++k) {
      float hv = tx * w1[2 * k] + ty * w1[2 * k + 1] + b1[k];
      float g = gelu_exact(hv);
      a0 += g * w2[k];
      a1 += g * w2[512 + k];
      a2 += g * w2[1024 + k];
    }
    tab[e * 3 + 0] = a0; tab[e * 3 + 1] = a1; tab[e * 3 + 2] = a2;
  }
  __syncthreads();
  for (int idx = tid; idx < 3 * 49 * 49; idx += 256) {
    int h = idx / 2401, rem = idx % 2401;
    int n = rem / 49, m = rem % 49;
    int dr = n / 7 - m / 7 + 6;
    int dc = n % 7 - m % 7 + 6;
    float v = tab[(dr * 13 + dc) * 3 + h];
    bias16[idx] = 16.f / (1.f + __expf(-v));
  }
}

// ---------------- fused window attention (QKV + cosine attn + proj + LN1 + residual) ----------------
__global__ __launch_bounds__(128)
void swin_attn_kernel(const float* __restrict__ x,
                      const float* __restrict__ attn_mask,
                      const bf16_t* __restrict__ wqkv,
                      const float* __restrict__ q_bias,
                      const float* __restrict__ v_bias,
                      const float* __restrict__ logit_scale,
                      const float* __restrict__ bias16,
                      const bf16_t* __restrict__ wproj,
                      const float* __restrict__ proj_b,
                      const float* __restrict__ n1g,
                      const float* __restrict__ n1b,
                      float* __restrict__ x1) {
  __shared__ __align__(32) unsigned char smem[61440];
  bf16_t* xbf   = (bf16_t*)(smem);           // [64][96] bf16 (QKV phase only)
  bf16_t* attnA = (bf16_t*)(smem);           // [3][64][64] bf16 (overlays xbf, after QKV)
  bf16_t* qnA   = (bf16_t*)(smem + 24576);   // [3][64][32]
  bf16_t* knB   = (bf16_t*)(smem + 36864);   // [3][64][32]
  bf16_t* vT    = (bf16_t*)(smem + 49152);   // [3][32][64]
  bf16_t* outA  = (bf16_t*)(smem + 24576);   // [64][96] (overlays qnA, after scores)
  float*  projS = (float*)(smem + 36864);    // [64][96] f32 (overlays knB+vT, after attn@v)

  int tid = threadIdx.x;
  int lane = tid & 31;
  int wv = tid >> 5;
  int win = blockIdx.x;
  int b = win >> 6;
  int wl = win & 63;
  int whh = wl >> 3;
  int www = wl & 7;

  // gather shifted window -> bf16, pad rows 49..63 with zeros
  for (int i = tid; i < 64 * 96; i += 128) {
    int t = i / 96, c = i % 96;
    float v = 0.f;
    if (t < NTOK) {
      int r = t / 7, cc = t % 7;
      int gh = (whh * 7 + r + 3) % HW_;
      int gw = (www * 7 + cc + 3) % HW_;
      v = x[((size_t)(b * 3136 + gh * HW_ + gw)) * C_ + c];
    }
    xbf[i] = (bf16_t)v;
  }
  __syncthreads();

  const float ln100 = 4.60517018598809f;
  int mBase = wv * 16;
  int cl = lane & 15;
  int halfOff = (lane >> 4) << 3;

  // ---- QKV: (64x96) @ (288x96)^T, per wave one m-tile, 9 (set,head) pairs ----
  for (int idx = 0; idx < 9; ++idx) {
    int set = idx / 3, h = idx % 3;
    v8f a0 = zero8(), a1 = zero8();
#pragma unroll
    for (int kt = 0; kt < 3; ++kt) {
      v16bf aF = load_fragA(xbf, mBase, 96, kt * 32, lane);
      v16bf bF0 = load_fragB(wqkv, (2 * idx) * 16, 96, kt * 32, lane);
      v16bf bF1 = load_fragB(wqkv, (2 * idx + 1) * 16, 96, kt * 32, lane);
      a0 = wmma_bf16(aF, bF0, a0);
      a1 = wmma_bf16(aF, bF1, a1);
    }
    int n0 = idx * 32 + cl;
    float bi0 = 0.f, bi1 = 0.f;
    if (set == 0) { bi0 = q_bias[n0]; bi1 = q_bias[n0 + 16]; }
    else if (set == 2) { bi0 = v_bias[n0 - 192]; bi1 = v_bias[n0 - 176]; }
#pragma unroll
    for (int r = 0; r < 8; ++r) { a0[r] += bi0; a1[r] += bi1; }
    int mh = mBase + halfOff;
    if (set < 2) {
      float scl = 1.f;
      if (set == 0) scl = __expf(fminf(logit_scale[h], ln100));
#pragma unroll
      for (int r = 0; r < 8; ++r) {
        float ss = a0[r] * a0[r] + a1[r] * a1[r];
        ss += __shfl_xor(ss, 1, 32);
        ss += __shfl_xor(ss, 2, 32);
        ss += __shfl_xor(ss, 4, 32);
        ss += __shfl_xor(ss, 8, 32);
        float rs = scl / fmaxf(sqrtf(ss), 1e-12f);
        a0[r] *= rs; a1[r] *= rs;
      }
      bf16_t* dst = (set == 0 ? qnA : knB) + h * 64 * 32;
#pragma unroll
      for (int r = 0; r < 8; ++r) {
        int m = mh + r;
        dst[m * 32 + cl] = (bf16_t)a0[r];
        dst[m * 32 + 16 + cl] = (bf16_t)a1[r];
      }
    } else {
      bf16_t* dst = vT + h * 32 * 64;   // [d][key]
#pragma unroll
      for (int r = 0; r < 8; ++r) {
        int m = mh + r;
        dst[cl * 64 + m] = (bf16_t)a0[r];
        dst[(16 + cl) * 64 + m] = (bf16_t)a1[r];
      }
    }
  }
  __syncthreads();

  // ---- scores = qn@kn^T (+bias+mask) + softmax -> attnA (bf16) ----
  for (int s = 0; s < 3; ++s) {
    int sid = wv + 4 * s;           // 12 strips: (head, m-tile)
    int h = sid >> 2, mt = sid & 3;
    v8f sa[4];
#pragma unroll
    for (int nt = 0; nt < 4; ++nt) {
      v16bf aF = load_fragA(qnA + h * 2048, mt * 16, 32, 0, lane);
      v16bf bF = load_fragB(knB + h * 2048, nt * 16, 32, 0, lane);
      sa[nt] = wmma_bf16(aF, bF, zero8());
    }
    int mh = mt * 16 + halfOff;
#pragma unroll
    for (int nt = 0; nt < 4; ++nt) {
      int col = nt * 16 + cl;
      bool cv = (col < NTOK);
      int cb = cv ? col : 0;
#pragma unroll
      for (int r = 0; r < 8; ++r) {
        int row = mh + r;
        int rb = row < NTOK ? row : 0;
        float vv = sa[nt][r] + bias16[(h * 49 + rb) * 49 + cb]
                 + attn_mask[(wl * 49 + rb) * 49 + cb];
        sa[nt][r] = cv ? vv : -1e30f;
      }
    }
#pragma unroll
    for (int r = 0; r < 8; ++r) {
      float mx = fmaxf(fmaxf(sa[0][r], sa[1][r]), fmaxf(sa[2][r], sa[3][r]));
      mx = fmaxf(mx, __shfl_xor(mx, 1, 32));
      mx = fmaxf(mx, __shfl_xor(mx, 2, 32));
      mx = fmaxf(mx, __shfl_xor(mx, 4, 32));
      mx = fmaxf(mx, __shfl_xor(mx, 8, 32));
      float e0 = __expf(sa[0][r] - mx), e1 = __expf(sa[1][r] - mx);
      float e2 = __expf(sa[2][r] - mx), e3 = __expf(sa[3][r] - mx);
      float sm = e0 + e1 + e2 + e3;
      sm += __shfl_xor(sm, 1, 32);
      sm += __shfl_xor(sm, 2, 32);
      sm += __shfl_xor(sm, 4, 32);
      sm += __shfl_xor(sm, 8, 32);
      float inv = 1.f / sm;
      sa[0][r] = e0 * inv; sa[1][r] = e1 * inv; sa[2][r] = e2 * inv; sa[3][r] = e3 * inv;
    }
    bf16_t* dA = attnA + h * 4096;
#pragma unroll
    for (int nt = 0; nt < 4; ++nt) {
      int col = nt * 16 + cl;
#pragma unroll
      for (int r = 0; r < 8; ++r) dA[(mh + r) * 64 + col] = (bf16_t)sa[nt][r];
    }
  }
  __syncthreads();

  // ---- out = attn @ v -> outA [64][96] bf16 ----
  for (int s = 0; s < 3; ++s) {
    int sid = wv + 4 * s;
    int h = sid >> 2, mt = sid & 3;
    v8f o0 = zero8(), o1 = zero8();
#pragma unroll
    for (int kt = 0; kt < 2; ++kt) {
      v16bf aF = load_fragA(attnA + h * 4096, mt * 16, 64, kt * 32, lane);
      v16bf bF0 = load_fragB(vT + h * 2048, 0, 64, kt * 32, lane);
      v16bf bF1 = load_fragB(vT + h * 2048, 16, 64, kt * 32, lane);
      o0 = wmma_bf16(aF, bF0, o0);
      o1 = wmma_bf16(aF, bF1, o1);
    }
    int mh = mt * 16 + halfOff;
#pragma unroll
    for (int r = 0; r < 8; ++r) {
      outA[(mh + r) * 96 + h * 32 + cl] = (bf16_t)o0[r];
      outA[(mh + r) * 96 + h * 32 + 16 + cl] = (bf16_t)o1[r];
    }
  }
  __syncthreads();

  // ---- proj: (64x96) @ (96x96)^T -> projS f32 ----
  {
    v8f p[6];
#pragma unroll
    for (int nt = 0; nt < 6; ++nt) p[nt] = zero8();
#pragma unroll
    for (int kt = 0; kt < 3; ++kt) {
      v16bf aF = load_fragA(outA, mBase, 96, kt * 32, lane);
#pragma unroll
      for (int nt = 0; nt < 6; ++nt) {
        v16bf bF = load_fragB(wproj, nt * 16, 96, kt * 32, lane);
        p[nt] = wmma_bf16(aF, bF, p[nt]);
      }
    }
    int mh = mBase + halfOff;
#pragma unroll
    for (int nt = 0; nt < 6; ++nt) {
      float pb = proj_b[nt * 16 + cl];
#pragma unroll
      for (int r = 0; r < 8; ++r)
        projS[(mh + r) * 96 + nt * 16 + cl] = p[nt][r] + pb;
    }
  }
  __syncthreads();

  // ---- LN1 + residual, scatter to natural layout ----
  for (int t = wv; t < NTOK; t += 4) {
    int c0 = lane * 3;
    float v0 = projS[t * 96 + c0];
    float v1 = projS[t * 96 + c0 + 1];
    float v2 = projS[t * 96 + c0 + 2];
    float s = v0 + v1 + v2;
    float q = v0 * v0 + v1 * v1 + v2 * v2;
#pragma unroll
    for (int m = 1; m < 32; m <<= 1) { s += __shfl_xor(s, m, 32); q += __shfl_xor(q, m, 32); }
    float mu = s * (1.f / 96.f);
    float var = q * (1.f / 96.f) - mu * mu;
    float rs = rsqrtf(var + 1e-5f);
    int r = t / 7, cc = t % 7;
    int gh = (whh * 7 + r + 3) % HW_;
    int gw = (www * 7 + cc + 3) % HW_;
    size_t tok = (size_t)(b * 3136 + gh * HW_ + gw);
    float vs[3] = {v0, v1, v2};
#pragma unroll
    for (int j = 0; j < 3; ++j) {
      int c = c0 + j;
      float xr = x[tok * C_ + c];
      x1[tok * C_ + c] = xr + (vs[j] - mu) * rs * n1g[c] + n1b[c];
    }
  }
}

// ---------------- fused MLP (fc1 + GELU + fc2 + LN2 + residual) ----------------
__global__ __launch_bounds__(128)
void swin_mlp_kernel(const float* __restrict__ x1,
                     const bf16_t* __restrict__ w1,
                     const float* __restrict__ b1,
                     const bf16_t* __restrict__ w2,
                     const float* __restrict__ b2,
                     const float* __restrict__ n2g,
                     const float* __restrict__ n2b,
                     float* __restrict__ out) {
  __shared__ __align__(32) bf16_t xA[64 * 96];    // 12 KB
  __shared__ __align__(32) bf16_t hA[64 * 384];   // 48 KB
  int tid = threadIdx.x;
  int lane = tid & 31;
  int wv = tid >> 5;
  size_t rowBase = (size_t)blockIdx.x * 64;

  for (int i = tid; i < 64 * 96; i += 128) xA[i] = (bf16_t)x1[rowBase * 96 + i];
  __syncthreads();

  int mBase = wv * 16;
  int cl = lane & 15;
  int halfOff = (lane >> 4) << 3;
  int mh = mBase + halfOff;

  // fc1 + GELU -> hA bf16
  for (int nt = 0; nt < 24; ++nt) {
    v8f a = zero8();
#pragma unroll
    for (int kt = 0; kt < 3; ++kt) {
      v16bf aF = load_fragA(xA, mBase, 96, kt * 32, lane);
      v16bf bF = load_fragB(w1, nt * 16, 96, kt * 32, lane);
      a = wmma_bf16(aF, bF, a);
    }
    int col = nt * 16 + cl;
    float bb = b1[col];
#pragma unroll
    for (int r = 0; r < 8; ++r) {
      float v = a[r] + bb;
      hA[(mh + r) * 384 + col] = (bf16_t)gelu_exact(v);
    }
  }
  __syncthreads();

  // fc2, keep all 6 n-tiles resident
  v8f acc[6];
#pragma unroll
  for (int nt = 0; nt < 6; ++nt) acc[nt] = zero8();
  for (int kt = 0; kt < 12; ++kt) {
    v16bf aF = load_fragA(hA, mBase, 384, kt * 32, lane);
#pragma unroll
    for (int nt = 0; nt < 6; ++nt) {
      v16bf bF = load_fragB(w2, nt * 16, 384, kt * 32, lane);
      acc[nt] = wmma_bf16(aF, bF, acc[nt]);
    }
  }
#pragma unroll
  for (int nt = 0; nt < 6; ++nt) {
    float bb = b2[nt * 16 + cl];
#pragma unroll
    for (int r = 0; r < 8; ++r) acc[nt][r] += bb;
  }

  // LN2 + residual, in-register reductions per row (16 lanes x 6 tiles = 96 cols)
#pragma unroll
  for (int r = 0; r < 8; ++r) {
    float s = 0.f, q = 0.f;
#pragma unroll
    for (int nt = 0; nt < 6; ++nt) { float v = acc[nt][r]; s += v; q += v * v; }
    s += __shfl_xor(s, 1, 32); q += __shfl_xor(q, 1, 32);
    s += __shfl_xor(s, 2, 32); q += __shfl_xor(q, 2, 32);
    s += __shfl_xor(s, 4, 32); q += __shfl_xor(q, 4, 32);
    s += __shfl_xor(s, 8, 32); q += __shfl_xor(q, 8, 32);
    float mu = s * (1.f / 96.f);
    float var = q * (1.f / 96.f) - mu * mu;
    float rs = rsqrtf(var + 1e-5f);
    size_t tok = rowBase + (size_t)(mh + r);
#pragma unroll
    for (int nt = 0; nt < 6; ++nt) {
      int c = nt * 16 + cl;
      out[tok * 96 + c] = x1[tok * 96 + c] + (acc[nt][r] - mu) * rs * n2g[c] + n2b[c];
    }
  }
}

// ---------------- launcher ----------------
extern "C" void kernel_launch(void* const* d_in, const int* in_sizes, int n_in,
                              void* d_out, int out_size, void* d_ws, size_t ws_size,
                              hipStream_t stream) {
  const float* x        = (const float*)d_in[0];
  const float* mask     = (const float*)d_in[1];
  const float* qkv_w    = (const float*)d_in[2];
  const float* q_bias   = (const float*)d_in[3];
  const float* v_bias   = (const float*)d_in[4];
  const float* lscale   = (const float*)d_in[5];
  const float* cpb_w1   = (const float*)d_in[6];
  const float* cpb_b1   = (const float*)d_in[7];
  const float* cpb_w2   = (const float*)d_in[8];
  const float* proj_w   = (const float*)d_in[9];
  const float* proj_b   = (const float*)d_in[10];
  const float* n1g      = (const float*)d_in[11];
  const float* n1b      = (const float*)d_in[12];
  const float* fc1_w    = (const float*)d_in[13];
  const float* fc1_b    = (const float*)d_in[14];
  const float* fc2_w    = (const float*)d_in[15];
  const float* fc2_b    = (const float*)d_in[16];
  const float* n2g      = (const float*)d_in[17];
  const float* n2b      = (const float*)d_in[18];

  unsigned char* ws = (unsigned char*)d_ws;
  bf16_t* wq   = (bf16_t*)(ws + 0);         // 288*96*2  = 55296
  bf16_t* wp   = (bf16_t*)(ws + 55296);     // 96*96*2   = 18432
  bf16_t* w1b  = (bf16_t*)(ws + 73728);     // 384*96*2  = 73728
  bf16_t* w2b  = (bf16_t*)(ws + 147456);    // 96*384*2  = 73728
  float*  bias16 = (float*)(ws + 221184);   // 3*49*49*4 = 28812
  float*  x1   = (float*)(ws + 262144);     // 200704*96*4 = 77070336

  convert_weights_kernel<<<64, 256, 0, stream>>>(qkv_w, proj_w, fc1_w, fc2_w, wq, wp, w1b, w2b);
  cpb_bias_kernel<<<1, 256, 0, stream>>>(cpb_w1, cpb_b1, cpb_w2, bias16);
  swin_attn_kernel<<<4096, 128, 0, stream>>>(x, mask, wq, q_bias, v_bias, lscale,
                                             bias16, wp, proj_b, n1g, n1b, x1);
  swin_mlp_kernel<<<3136, 128, 0, stream>>>(x1, w1b, fc1_b, w2b, fc2_b, n2g, n2b,
                                            (float*)d_out);
}
